// WavenetModel_40931038331203
// MI455X (gfx1250) — matrix operand order repaired and checked by reading, not compiled, not run on hardware
//
#include <hip/hip_runtime.h>

typedef _Float16 f16;
typedef __attribute__((ext_vector_type(16))) _Float16 v16h;
typedef __attribute__((ext_vector_type(8)))  _Float16 v8h;
typedef __attribute__((ext_vector_type(8)))  float    v8f;

#define T_X      16384
#define T_FINAL  13315
#define NBATCH   16
#define NLAYER   30
#define KZ       (NLAYER * 32)   // 960: K of the fused skip GEMM

// ---- workspace layout (bytes) ----
// x buffers are time-major: [b][t][32ch] f32
#define OFF_XA   0ull
#define OFF_XB   33554432ull                       // 16*16384*32*4
#define OFF_Z    67108864ull                       // zall: [b][T_FINAL][960] f16 = 409,036,800 B
#define OFF_WF   (OFF_Z + 409036800ull)            // filt frags: 30*2*2*512 halves
#define OFF_WG   (OFF_WF + 122880ull)
#define OFF_WR   (OFF_WG + 122880ull)              // res frags: 30*2*512 halves
#define OFF_WS   (OFF_WR + 61440ull)               // skip frags: 16*30*512 halves
#define OFF_WE1  (OFF_WS + 491520ull)              // end1 frags: 16*8*512 halves
#define OFF_WE2  (OFF_WE1 + 131072ull)

// A-fragment K index, contiguous k order used here for A and B (16x32):
// lanes 0-15 hold K {0..15}, lanes 16-31 hold K {16..31}
__device__ __forceinline__ int kidxA(int lane, int h) {
  return h + ((lane >> 4) ? 16 : 0);
}

#define WMMA(A, B, C) __builtin_amdgcn_wmma_f32_16x16x32_f16(false, (A), false, (B), (short)0, (C), false, false)

__device__ __forceinline__ float fast_sig(float x) {
  return __builtin_amdgcn_rcpf(1.0f + __expf(-x));
}
__device__ __forceinline__ float fast_tanh(float x) {
  float e2 = __expf(2.0f * x);
  return (e2 - 1.0f) * __builtin_amdgcn_rcpf(e2 + 1.0f);
}

// -------------------- prologue kernels --------------------

// x0[b][t][c] = causal_w[c] * y[b][0][t] + causal_b[c]   (time-major)
__global__ void causal_kernel(const float* __restrict__ y,
                              const float* __restrict__ cw,
                              const float* __restrict__ cb,
                              float* __restrict__ x0) {
  int i = blockIdx.x * blockDim.x + threadIdx.x;
  if (i >= NBATCH * T_X * 32) return;
  int c = i & 31;
  int t = (i >> 5) & (T_X - 1);
  int b = i >> 19;
  x0[i] = cw[c] * y[b * T_X + t] + cb[c];
}

// filt/gate: [30][32co][32ci][2tap] -> frags [30][tap][mt][32 lanes][16 halves]
__global__ void swz_fg_kernel(const float* __restrict__ W, f16* __restrict__ out) {
  int tid = blockIdx.x * blockDim.x + threadIdx.x;
  if (tid >= 30 * 2 * 2 * 32) return;
  int lane = tid & 31, mt = (tid >> 5) & 1, tap = (tid >> 6) & 1, l = tid >> 7;
  int co = mt * 16 + (lane & 15);
  v16h v;
#pragma unroll
  for (int h = 0; h < 16; ++h)
    v[h] = (f16)W[((l * 32 + co) * 32 + kidxA(lane, h)) * 2 + tap];
  ((v16h*)out)[(l * 4 + tap * 2 + mt) * 32 + lane] = v;
}

// res: [30][32co][32ci] -> frags [30][mt][32][16]
__global__ void swz_res_kernel(const float* __restrict__ W, f16* __restrict__ out) {
  int tid = blockIdx.x * blockDim.x + threadIdx.x;
  if (tid >= 30 * 2 * 32) return;
  int lane = tid & 31, mt = (tid >> 5) & 1, l = tid >> 6;
  int co = mt * 16 + (lane & 15);
  v16h v;
#pragma unroll
  for (int h = 0; h < 16; ++h) v[h] = (f16)W[(l * 32 + co) * 32 + kidxA(lane, h)];
  ((v16h*)out)[(l * 2 + mt) * 32 + lane] = v;
}

// skip: [30][256co][32ci] -> big-K frags [16mt][30kb(=layer)][32][16]
__global__ void swz_skip_kernel(const float* __restrict__ W, f16* __restrict__ out) {
  int tid = blockIdx.x * blockDim.x + threadIdx.x;
  if (tid >= 16 * 30 * 32) return;
  int lane = tid & 31, kb = (tid >> 5) % 30, mt = tid / (30 * 32);
  int co = mt * 16 + (lane & 15);
  v16h v;
#pragma unroll
  for (int h = 0; h < 16; ++h) v[h] = (f16)W[(kb * 256 + co) * 32 + kidxA(lane, h)];
  ((v16h*)out)[(mt * 30 + kb) * 32 + lane] = v;
}

// end1/end2: [256co][256ci] -> frags [16mt][8kb][32][16]
__global__ void swz_end_kernel(const float* __restrict__ W, f16* __restrict__ out) {
  int tid = blockIdx.x * blockDim.x + threadIdx.x;
  if (tid >= 16 * 8 * 32) return;
  int lane = tid & 31, kb = (tid >> 5) & 7, mt = tid >> 8;
  int co = mt * 16 + (lane & 15);
  v16h v;
#pragma unroll
  for (int h = 0; h < 16; ++h) v[h] = (f16)W[co * 256 + kb * 32 + kidxA(lane, h)];
  ((v16h*)out)[(mt * 8 + kb) * 32 + lane] = v;
}

// -------------------- per-layer fused kernel --------------------
// x is time-major [b][t][32] f32. One wave32 per 16-column tile:
//   f/g = Wf/Wg (2 taps) @ x  (8 WMMA) -> z = tanh(f)*sigmoid(g)
//   z -> zall[b][t][l*32..] (f16, feeds the fused K=960 skip GEMM in the head)
//   z restaged via LDS transpose -> B fragment
//   x_out = Wr @ z + residual (2 WMMA)
__global__ __launch_bounds__(256)
void layer_kernel(const float* __restrict__ x_in, float* __restrict__ x_out,
                  f16* __restrict__ zall,
                  const f16* __restrict__ wf, const f16* __restrict__ wg,
                  const f16* __restrict__ wr,
                  int Tin, int d, int skip_off, int lbase) {
  __shared__ f16 zT[8][16][32];  // [wave][time col][ci]
  const int Tout = Tin - d;
  const int wave = threadIdx.x >> 5;
  const int lane = threadIdx.x & 31;
  const int t0 = (blockIdx.x * 8 + wave) * 16;
  if (t0 >= Tout) return;
  const int b = blockIdx.y;
  const int n = lane & 15;
  const int hi = lane >> 4;

  const float* xb = x_in + (size_t)b * T_X * 32;
  int tc0 = t0 + n;      if (tc0 > Tin - 1) tc0 = Tin - 1;
  int tc1 = t0 + n + d;  if (tc1 > Tin - 1) tc1 = Tin - 1;

  // B fragments: 16 contiguous channels at one timestep (vectorizable loads)
  const float* p0 = xb + (size_t)tc0 * 32 + hi * 16;
  const float* p1 = xb + (size_t)tc1 * 32 + hi * 16;
  v16h b0, b1;
#pragma unroll
  for (int h = 0; h < 16; ++h) {
    b0[h] = (f16)p0[h];
    b1[h] = (f16)p1[h];
  }

  const v16h* F = (const v16h*)wf;
  const v16h* G = (const v16h*)wg;
  v8f f0 = {}, f1 = {}, g0 = {}, g1 = {};
  f0 = WMMA(F[(0 * 2 + 0) * 32 + lane], b0, f0);
  f0 = WMMA(F[(1 * 2 + 0) * 32 + lane], b1, f0);
  f1 = WMMA(F[(0 * 2 + 1) * 32 + lane], b0, f1);
  f1 = WMMA(F[(1 * 2 + 1) * 32 + lane], b1, f1);
  g0 = WMMA(G[(0 * 2 + 0) * 32 + lane], b0, g0);
  g0 = WMMA(G[(1 * 2 + 0) * 32 + lane], b1, g0);
  g1 = WMMA(G[(0 * 2 + 1) * 32 + lane], b0, g1);
  g1 = WMMA(G[(1 * 2 + 1) * 32 + lane], b1, g1);

  // z = tanh(f) * sigmoid(g)  (branchless exp/rcp forms)
  v8h z0, z1;
#pragma unroll
  for (int r = 0; r < 8; ++r) {
    z0[r] = (f16)(fast_tanh(f0[r]) * fast_sig(g0[r]));
    z1[r] = (f16)(fast_tanh(f1[r]) * fast_sig(g1[r]));
  }

  const bool col_ok = (t0 + n) < Tout;

  // write z into the concatenated activation tensor (final-window aligned)
  int sc = t0 + n - skip_off;
  if (col_ok && sc >= 0) {
    f16* zp = zall + ((size_t)b * T_FINAL + sc) * KZ + lbase;
    *(v8h*)&zp[hi * 8]      = z0;
    *(v8h*)&zp[16 + hi * 8] = z1;
  }

  // restage z (transposed) in LDS to build the B fragment for the res GEMM
  *(v8h*)&zT[wave][n][hi * 8]      = z0;
  *(v8h*)&zT[wave][n][16 + hi * 8] = z1;
  __asm__ volatile("s_wait_dscnt 0" ::: "memory");  // intra-wave LDS RAW
  v16h bz = *(const v16h*)&zT[wave][n][hi * 16];

  // residual path: x_out[t] = Wr @ z + x_in[t+d]
  const v16h* R = (const v16h*)wr;
  v8f r0 = {}, r1 = {};
  r0 = WMMA(R[0 * 32 + lane], bz, r0);
  r1 = WMMA(R[1 * 32 + lane], bz, r1);
  if (col_ok) {
    const float* rp = xb + (size_t)tc1 * 32;
    float* xo = x_out + ((size_t)b * T_X + (t0 + n)) * 32;
#pragma unroll
    for (int r = 0; r < 8; ++r) {
      xo[hi * 8 + r]      = r0[r] + rp[hi * 8 + r];
      xo[16 + hi * 8 + r] = r1[r] + rp[16 + hi * 8 + r];
    }
  }
}

// -------------------- fused head --------------------
// skip GEMM (256 x 960, K over all 30 layers' z) -> relu
//   -> end1 GEMM (256x256) + bias -> relu -> end2 GEMM (256x256) + bias
__global__ __launch_bounds__(128)
void head_kernel(const f16* __restrict__ zall, float* __restrict__ out,
                 const f16* __restrict__ wsk,
                 const f16* __restrict__ we1, const f16* __restrict__ we2,
                 const float* __restrict__ bias1, const float* __restrict__ bias2) {
  __shared__ f16 sT[4][16][256];  // [wave][time col][channel], reused between stages
  const int wave = threadIdx.x >> 5;
  const int lane = threadIdx.x & 31;
  const int t0 = (blockIdx.x * 4 + wave) * 16;
  if (t0 >= T_FINAL) return;
  const int b = blockIdx.y;
  const int n = lane & 15;
  const int hi = lane >> 4;

  int tc = t0 + n; if (tc > T_FINAL - 1) tc = T_FINAL - 1;
  const f16* zp = zall + ((size_t)b * T_FINAL + tc) * KZ;

  const v16h* SK = (const v16h*)wsk;
  const v16h* E1 = (const v16h*)we1;
  const v16h* E2 = (const v16h*)we2;

  v8f acc[16];
#pragma unroll
  for (int mt = 0; mt < 16; ++mt) acc[mt] = (v8f){};

  // fused skip GEMM: K = 960, B fragments are contiguous f16 (no conversion)
  for (int kb = 0; kb < NLAYER; ++kb) {
    v16h bz = *(const v16h*)&zp[kb * 32 + hi * 16];
#pragma unroll
    for (int mt = 0; mt < 16; ++mt)
      acc[mt] = WMMA(SK[(mt * NLAYER + kb) * 32 + lane], bz, acc[mt]);
  }

  // relu -> stage for end1
#pragma unroll
  for (int mt = 0; mt < 16; ++mt) {
    v8h p;
#pragma unroll
    for (int r = 0; r < 8; ++r) {
      float v = acc[mt][r];
      p[r] = (f16)(v > 0.0f ? v : 0.0f);
    }
    *(v8h*)&sT[wave][n][mt * 16 + hi * 8] = p;
  }
  __asm__ volatile("s_wait_dscnt 0" ::: "memory");

  // end1 GEMM
#pragma unroll
  for (int mt = 0; mt < 16; ++mt) acc[mt] = (v8f){};
  for (int kb = 0; kb < 8; ++kb) {
    v16h bh = *(const v16h*)&sT[wave][n][kb * 32 + hi * 16];
#pragma unroll
    for (int mt = 0; mt < 16; ++mt)
      acc[mt] = WMMA(E1[(mt * 8 + kb) * 32 + lane], bh, acc[mt]);
  }

  // + bias, relu -> restage (LDS ops are wave-ordered, safe to reuse sT)
#pragma unroll
  for (int mt = 0; mt < 16; ++mt) {
    v8h p;
#pragma unroll
    for (int r = 0; r < 8; ++r) {
      int co = mt * 16 + hi * 8 + r;
      float v = acc[mt][r] + bias1[co];
      p[r] = (f16)(v > 0.0f ? v : 0.0f);
    }
    *(v8h*)&sT[wave][n][mt * 16 + hi * 8] = p;
  }
  __asm__ volatile("s_wait_dscnt 0" ::: "memory");

  // end2 GEMM
#pragma unroll
  for (int mt = 0; mt < 16; ++mt) acc[mt] = (v8f){};
  for (int kb = 0; kb < 8; ++kb) {
    v16h bh = *(const v16h*)&sT[wave][n][kb * 32 + hi * 16];
#pragma unroll
    for (int mt = 0; mt < 16; ++mt)
      acc[mt] = WMMA(E2[(mt * 8 + kb) * 32 + lane], bh, acc[mt]);
  }

  if (t0 + n < T_FINAL) {
    float* ob = out + (size_t)b * 256 * T_FINAL;
#pragma unroll
    for (int mt = 0; mt < 16; ++mt) {
#pragma unroll
      for (int r = 0; r < 8; ++r) {
        int co = mt * 16 + hi * 8 + r;
        ob[(size_t)co * T_FINAL + t0 + n] = acc[mt][r] + bias2[co];
      }
    }
  }
}

// -------------------- host --------------------

extern "C" void kernel_launch(void* const* d_in, const int* in_sizes, int n_in,
                              void* d_out, int out_size, void* d_ws, size_t ws_size,
                              hipStream_t stream) {
  const float* y        = (const float*)d_in[0];
  const float* causal_w = (const float*)d_in[1];
  const float* causal_b = (const float*)d_in[2];
  const float* filt_w   = (const float*)d_in[3];
  const float* gate_w   = (const float*)d_in[4];
  const float* res_w    = (const float*)d_in[5];
  const float* skip_w   = (const float*)d_in[6];
  const float* end1_w   = (const float*)d_in[7];
  const float* end1_b   = (const float*)d_in[8];
  const float* end2_w   = (const float*)d_in[9];
  const float* end2_b   = (const float*)d_in[10];

  char* ws = (char*)d_ws;
  float* xA   = (float*)(ws + OFF_XA);
  float* xB   = (float*)(ws + OFF_XB);
  f16*   zall = (f16*)(ws + OFF_Z);
  f16*   wf   = (f16*)(ws + OFF_WF);
  f16*   wg   = (f16*)(ws + OFF_WG);
  f16*   wr   = (f16*)(ws + OFF_WR);
  f16*   wsk  = (f16*)(ws + OFF_WS);
  f16*   we1  = (f16*)(ws + OFF_WE1);
  f16*   we2  = (f16*)(ws + OFF_WE2);
  float* outf = (float*)d_out;

  // causal 1x1 conv (1 -> 32), time-major output
  {
    int ntot = NBATCH * T_X * 32;
    causal_kernel<<<(ntot + 255) / 256, 256, 0, stream>>>(y, causal_w, causal_b, xA);
  }
  // weight swizzles into WMMA A-fragment layout (f16)
  swz_fg_kernel<<<(3840 + 255) / 256, 256, 0, stream>>>(filt_w, wf);
  swz_fg_kernel<<<(3840 + 255) / 256, 256, 0, stream>>>(gate_w, wg);
  swz_res_kernel<<<(1920 + 255) / 256, 256, 0, stream>>>(res_w, wr);
  swz_skip_kernel<<<(15360 + 255) / 256, 256, 0, stream>>>(skip_w, wsk);
  swz_end_kernel<<<(4096 + 255) / 256, 256, 0, stream>>>(end1_w, we1);
  swz_end_kernel<<<(4096 + 255) / 256, 256, 0, stream>>>(end2_w, we2);

  // 30 dilated gated layers
  int Tin = T_X;
  int ping = 0;
  float* xb[2] = {xA, xB};
  for (int i = 0; i < NLAYER; ++i) {
    int d = 1 << (i % 10);
    int Tout = Tin - d;
    int tiles = (Tout + 15) / 16;
    dim3 grid((tiles + 7) / 8, NBATCH);
    layer_kernel<<<grid, 256, 0, stream>>>(
        xb[ping], xb[1 - ping], zall,
        wf + (size_t)i * 2048, wg + (size_t)i * 2048, wr + (size_t)i * 1024,
        Tin, d, Tout - T_FINAL, i * 32);
    ping ^= 1;
    Tin = Tout;
  }

  // fused head: skip(K=960) -> relu -> end1 -> relu -> end2
  {
    int tiles = (T_FINAL + 15) / 16;
    dim3 grid((tiles + 3) / 4, NBATCH);
    head_kernel<<<grid, 128, 0, stream>>>(zall, outf, wsk, we1, we2, end1_b, end2_b);
  }
}
